// HFGPT_35648228557143
// MI455X (gfx1250) — matrix operand order repaired and checked by reading, not compiled, not run on hardware
//
#include <hip/hip_runtime.h>

// ---------------------------------------------------------------------------
// Types for CDNA5 WMMA (wave32): v_wmma_f32_16x16x32_bf16
// ---------------------------------------------------------------------------
typedef __bf16 bf16x8  __attribute__((ext_vector_type(8)));
typedef __bf16 bf16x16 __attribute__((ext_vector_type(16)));
typedef float  f32x8   __attribute__((ext_vector_type(8)));

#define DEV __device__ __forceinline__

// Assemble a 16-element bf16 fragment from two contiguous 16B (ds_load_b128) loads.
DEV bf16x16 mk16(const __bf16* lo, const __bf16* hi) {
  bf16x8 a = *(const bf16x8*)lo;
  bf16x8 b = *(const bf16x8*)hi;
  bf16x16 r;
#pragma unroll
  for (int i = 0; i < 8; ++i) { r[i] = a[i]; r[i + 8] = b[i]; }
  return r;
}

DEV f32x8 wmma_bf16(bf16x16 a, bf16x16 b, f32x8 c) {
  // (neg_a, A, neg_b, B, c_mod, C, reuse_a, reuse_b)
  return __builtin_amdgcn_wmma_f32_16x16x32_bf16(false, a, false, b, (short)0, c,
                                                 false, false);
}

// Direct global -> LDS async copy of 16 bytes per lane (ASYNCcnt-tracked).
DEV void async_copy_b128(uint32_t lds_byte_off, const void* gaddr) {
  asm volatile("global_load_async_to_lds_b128 %0, %1, off"
               :: "v"(lds_byte_off), "v"(gaddr) : "memory");
}

DEV float gelu_new_f(float x) {
  const float c = 0.79788456080286535588f;  // sqrt(2/pi)
  float u = c * (x + 0.044715f * x * x * x);
  return 0.5f * x * (1.0f + tanhf(u));
}

// ---------------------------------------------------------------------------
// Weight transpose + fp32 -> bf16.  src: (R x C) fp32, dst: (C x R) bf16.
// grid.z = layer index; each layer offset R*C.
// ---------------------------------------------------------------------------
__global__ __launch_bounds__(256) void wtrans_kernel(const float* __restrict__ src,
                                                     __bf16* __restrict__ dst,
                                                     int R, int C) {
  __shared__ float tile[32][33];
  const size_t lo = (size_t)blockIdx.z * R * C;
  const int cb = blockIdx.x * 32, rb = blockIdx.y * 32;
  const int tx = threadIdx.x, ty = threadIdx.y;  // block (32, 8)
#pragma unroll
  for (int i = 0; i < 32; i += 8)
    tile[ty + i][tx] = src[lo + (size_t)(rb + ty + i) * C + cb + tx];
  __syncthreads();
#pragma unroll
  for (int i = 0; i < 32; i += 8)
    dst[lo + (size_t)(cb + ty + i) * R + rb + tx] = (__bf16)tile[tx][ty + i];
}

// ---------------------------------------------------------------------------
// h0[b][l][e] = x[l][b][e] + pos[l][e]; write fp32 + bf16
// ---------------------------------------------------------------------------
__global__ __launch_bounds__(256) void init_h_kernel(const float* __restrict__ x,
                                                     const float* __restrict__ pos,
                                                     float* __restrict__ hF,
                                                     __bf16* __restrict__ hB) {
  const int idx = blockIdx.x * 256 + threadIdx.x;  // over 8192*768
  const int e = idx % 768;
  const int bl = idx / 768;
  const int b = bl / 512;
  const int l = bl % 512;
  float v = x[((size_t)l * 16 + b) * 768 + e] + pos[(size_t)l * 768 + e];
  hF[idx] = v;
  hB[idx] = (__bf16)v;
}

// out[l][b][e] = h[b][l][e]
__global__ __launch_bounds__(256) void final_out_kernel(const float* __restrict__ hF,
                                                        float* __restrict__ out) {
  const int idx = blockIdx.x * 256 + threadIdx.x;  // over 512*16*768
  const int e = idx % 768;
  const int lb = idx / 768;
  const int b = lb % 16;
  const int l = lb / 16;
  out[idx] = hF[((size_t)b * 512 + l) * 768 + e];
}

// ---------------------------------------------------------------------------
// Fused residual + LayerNorm over E=768.  One block per row, 256 threads x 3.
// ---------------------------------------------------------------------------
__global__ __launch_bounds__(256) void resid_ln_kernel(
    const float* __restrict__ resid, const float* __restrict__ delta,
    const float* __restrict__ gamma, const float* __restrict__ beta,
    float* __restrict__ outF, __bf16* __restrict__ outB) {
  __shared__ float red[256];
  const int row = blockIdx.x, t = threadIdx.x;
  const size_t base = (size_t)row * 768;
  float x0 = resid[base + t] + delta[base + t];
  float x1 = resid[base + t + 256] + delta[base + t + 256];
  float x2 = resid[base + t + 512] + delta[base + t + 512];
  red[t] = x0 + x1 + x2;
  __syncthreads();
  for (int s = 128; s > 0; s >>= 1) {
    if (t < s) red[t] += red[t + s];
    __syncthreads();
  }
  float mu = red[0] * (1.0f / 768.0f);
  __syncthreads();
  float d0 = x0 - mu, d1 = x1 - mu, d2 = x2 - mu;
  red[t] = d0 * d0 + d1 * d1 + d2 * d2;
  __syncthreads();
  for (int s = 128; s > 0; s >>= 1) {
    if (t < s) red[t] += red[t + s];
    __syncthreads();
  }
  float inv = rsqrtf(red[0] * (1.0f / 768.0f) + 1e-5f);
  float y0 = d0 * inv * gamma[t] + beta[t];
  float y1 = d1 * inv * gamma[t + 256] + beta[t + 256];
  float y2 = d2 * inv * gamma[t + 512] + beta[t + 512];
  outF[base + t] = y0;        outB[base + t] = (__bf16)y0;
  outF[base + t + 256] = y1;  outB[base + t + 256] = (__bf16)y1;
  outF[base + t + 512] = y2;  outB[base + t + 512] = (__bf16)y2;
}

// ---------------------------------------------------------------------------
// WMMA GEMM: C(MxN) = A(MxK, bf16) * Bt(NxK, bf16)^T + bias, fused epilogue.
// Block tile 128x128, 8 waves (4 M-groups x 2 N-groups), wave tile 32x64,
// K-step 32 via v_wmma_f32_16x16x32_bf16.
// Double-buffered tile staging with GLOBAL_LOAD_ASYNC_TO_LDS_B128:
//   - each thread owns 4 x 16B chunks per stage (2 A + 2 B)
//   - barrier #1: the buffer we are about to overwrite is no longer read
//   - issue next stage's async copies, s_wait_asynccnt 4 retires current stage
//     (async loads complete in order), barrier #2 publishes LDS to all waves
// OP==0: fp32 store (+bias).  OP==1: gelu_new -> bf16 store (+bias).
// ---------------------------------------------------------------------------
template <int OP>
__global__ __launch_bounds__(256) void gemm_bf16_wmma(
    const __bf16* __restrict__ A, const __bf16* __restrict__ Bt,
    const float* __restrict__ bias, float* __restrict__ Cf,
    __bf16* __restrict__ Cb, int M, int N, int K) {
  __shared__ __bf16 As[2][128 * 32];
  __shared__ __bf16 Bs[2][128 * 32];
  const int tid = threadIdx.x;
  const int wave = tid >> 5, lane = tid & 31;
  const int g = lane >> 4, ln = lane & 15;
  const int wm = wave & 3, wn = wave >> 2;
  const int mBase = blockIdx.x * 128;
  const int nBase = blockIdx.y * 128;

  // this thread's two (row, k-chunk) staging assignments (512 chunks / 256 thr)
  const int r0 = tid >> 2, c0 = (tid & 3) * 8;
  const int r1 = (tid + 256) >> 2, c1 = ((tid + 256) & 3) * 8;
  const __bf16* gA0 = &A[(size_t)(mBase + r0) * K + c0];
  const __bf16* gA1 = &A[(size_t)(mBase + r1) * K + c1];
  const __bf16* gB0 = &Bt[(size_t)(nBase + r0) * K + c0];
  const __bf16* gB1 = &Bt[(size_t)(nBase + r1) * K + c1];

  auto issue_stage = [&](int buf, int k0) {
    async_copy_b128((uint32_t)(uintptr_t)&As[buf][r0 * 32 + c0], gA0 + k0);
    async_copy_b128((uint32_t)(uintptr_t)&As[buf][r1 * 32 + c1], gA1 + k0);
    async_copy_b128((uint32_t)(uintptr_t)&Bs[buf][r0 * 32 + c0], gB0 + k0);
    async_copy_b128((uint32_t)(uintptr_t)&Bs[buf][r1 * 32 + c1], gB1 + k0);
  };

  f32x8 acc[2][4];
#pragma unroll
  for (int s = 0; s < 2; ++s)
#pragma unroll
    for (int t = 0; t < 4; ++t) acc[s][t] = (f32x8)0.0f;

  const int steps = K / 32;
  issue_stage(0, 0);  // prologue prefetch

  for (int kk = 0; kk < steps; ++kk) {
    const int cur = kk & 1;
    __syncthreads();  // buffer cur^1 free for overwrite (prev compute done)
    if (kk + 1 < steps) {
      issue_stage(cur ^ 1, (kk + 1) * 32);
      asm volatile("s_wait_asynccnt 0x4" ::: "memory");  // current stage done
    } else {
      asm volatile("s_wait_asynccnt 0x0" ::: "memory");
    }
    __syncthreads();  // all waves' current-stage LDS writes visible

    bf16x16 af[2];
#pragma unroll
    for (int s = 0; s < 2; ++s) {
      const __bf16* rp = &As[cur][(wm * 32 + s * 16 + ln) * 32];
      af[s] = mk16(rp + g * 8, rp + 16 + g * 8);
    }
    bf16x16 bfr[4];
#pragma unroll
    for (int t = 0; t < 4; ++t) {
      const __bf16* rp = &Bs[cur][(wn * 64 + t * 16 + ln) * 32];
      bfr[t] = mk16(rp + g * 8, rp + 16 + g * 8);
    }
#pragma unroll
    for (int s = 0; s < 2; ++s)
#pragma unroll
      for (int t = 0; t < 4; ++t) acc[s][t] = wmma_bf16(af[s], bfr[t], acc[s][t]);
  }

#pragma unroll
  for (int s = 0; s < 2; ++s)
#pragma unroll
    for (int t = 0; t < 4; ++t) {
      const int col = nBase + wn * 64 + t * 16 + ln;
      const float bi = bias[col];
#pragma unroll
      for (int r = 0; r < 8; ++r) {
        const int row = mBase + wm * 32 + s * 16 + g * 8 + r;
        float v = acc[s][t][r] + bi;
        if (OP == 0) {
          Cf[(size_t)row * N + col] = v;
        } else {
          Cb[(size_t)row * N + col] = (__bf16)gelu_new_f(v);
        }
      }
    }
}

// ---------------------------------------------------------------------------
// Flash attention per (qblock=128 rows, head, batch).  8 waves x 16 q rows.
// S = (Q/8) K^T via WMMA; causal + (-10000) + padding mask; online softmax;
// O = P V via WMMA; writes bf16 output in (B*L, 768) layout.
// ---------------------------------------------------------------------------
__global__ __launch_bounds__(256) void attn_flash_kernel(
    const float* __restrict__ qkv,   // (B*L) x 2304 : [q | k | v], head-major d
    const float* __restrict__ mask,  // B x L
    __bf16* __restrict__ out) {      // (B*L) x 768
  __shared__ __bf16 Kt[128 * 64];        // [key][d]
  __shared__ __bf16 Vt[64 * 128];        // [d][key]  (V^T)
  __shared__ __bf16 Ps[8 * 16 * 128];    // per-wave 16x128 P strips
  const int qb = blockIdx.x, h = blockIdx.y, b = blockIdx.z;
  const int tid = threadIdx.x, wave = tid >> 5, lane = tid & 31;
  const int g = lane >> 4, ln = lane & 15;
  const int qBase = qb * 128;
  const int rowA = qBase + wave * 16 + ln;
  const size_t qoff = ((size_t)b * 512 + rowA) * 2304 + h * 64;

  // Q fragments (d 0..31 and 32..63), pre-scaled by 1/sqrt(64)
  bf16x16 qf[2];
#pragma unroll
  for (int f = 0; f < 2; ++f) {
#pragma unroll
    for (int i = 0; i < 8; ++i) {
      int dd = f * 32 + ((i < 4) ? 0 : 16) + g * 8 + (i & 3) * 2;
      qf[f][2 * i] = (__bf16)(qkv[qoff + dd] * 0.125f);
      qf[f][2 * i + 1] = (__bf16)(qkv[qoff + dd + 1] * 0.125f);
    }
  }

  f32x8 O[4];
#pragma unroll
  for (int t = 0; t < 4; ++t) O[t] = (f32x8)0.0f;
  float mrun[8], lrun[8];
#pragma unroll
  for (int r = 0; r < 8; ++r) { mrun[r] = -3.0e38f; lrun[r] = 0.0f; }

  for (int j = 0; j <= qb; ++j) {  // causal: only key tiles <= query tile
    const int kBase = j * 128;
    __syncthreads();
    for (int idx = tid; idx < 128 * 64; idx += 256) {
      int key = idx >> 6, dd = idx & 63;
      size_t base = ((size_t)b * 512 + kBase + key) * 2304 + h * 64;
      Kt[key * 64 + dd] = (__bf16)qkv[base + 768 + dd];
      Vt[dd * 128 + key] = (__bf16)qkv[base + 1536 + dd];
    }
    __syncthreads();

    // S strip: 16 q rows x 128 keys
    f32x8 S[8];
#pragma unroll
    for (int t = 0; t < 8; ++t) S[t] = (f32x8)0.0f;
#pragma unroll
    for (int kk = 0; kk < 2; ++kk) {
#pragma unroll
      for (int t = 0; t < 8; ++t) {
        const __bf16* rp = &Kt[(t * 16 + ln) * 64 + kk * 32];
        bf16x16 bfrag = mk16(rp + g * 8, rp + 16 + g * 8);
        S[t] = wmma_bf16(qf[kk], bfrag, S[t]);
      }
    }

    // masking (causal -> exactly -10000, then additive padding mask)
    float rmax[8];
#pragma unroll
    for (int r = 0; r < 8; ++r) rmax[r] = -3.0e38f;
#pragma unroll
    for (int t = 0; t < 8; ++t) {
      const int kcol = kBase + t * 16 + ln;
      const float am = (1.0f - mask[b * 512 + kcol]) * -3.4028235e38f;
#pragma unroll
      for (int r = 0; r < 8; ++r) {
        const int qrow = qBase + wave * 16 + g * 8 + r;
        float v = (kcol <= qrow) ? S[t][r] : -10000.0f;
        v += am;
        S[t][r] = v;
        rmax[r] = fmaxf(rmax[r], v);
      }
    }
#pragma unroll
    for (int r = 0; r < 8; ++r)
#pragma unroll
      for (int off = 1; off < 16; off <<= 1)
        rmax[r] = fmaxf(rmax[r], __shfl_xor(rmax[r], off, 32));

    float lc[8];
#pragma unroll
    for (int r = 0; r < 8; ++r) {
      float mn = fmaxf(mrun[r], rmax[r]);
      lc[r] = __expf(mrun[r] - mn);
      mrun[r] = mn;
    }
    float psum[8];
#pragma unroll
    for (int r = 0; r < 8; ++r) psum[r] = 0.0f;
#pragma unroll
    for (int t = 0; t < 8; ++t)
#pragma unroll
      for (int r = 0; r < 8; ++r) {
        float p = __expf(S[t][r] - mrun[r]);
        S[t][r] = p;
        psum[r] += p;
      }
#pragma unroll
    for (int r = 0; r < 8; ++r) {
#pragma unroll
      for (int off = 1; off < 16; off <<= 1) psum[r] += __shfl_xor(psum[r], off, 32);
      lrun[r] = lrun[r] * lc[r] + psum[r];
    }
#pragma unroll
    for (int t = 0; t < 4; ++t)
#pragma unroll
      for (int r = 0; r < 8; ++r) O[t][r] *= lc[r];

    // stash P as bf16 in wave-private LDS strip
#pragma unroll
    for (int t = 0; t < 8; ++t)
#pragma unroll
      for (int r = 0; r < 8; ++r)
        Ps[(wave * 16 + g * 8 + r) * 128 + t * 16 + ln] = (__bf16)S[t][r];
    asm volatile("s_wait_dscnt 0" ::: "memory");  // wave-local LDS RAW fence

    // O += P (16x128) * V (128x64)
#pragma unroll
    for (int kk = 0; kk < 4; ++kk) {
      const __bf16* ap = &Ps[(wave * 16 + ln) * 128 + kk * 32];
      bf16x16 pa = mk16(ap + g * 8, ap + 16 + g * 8);
#pragma unroll
      for (int t = 0; t < 4; ++t) {
        const __bf16* vp = &Vt[(t * 16 + ln) * 128 + kk * 32];
        bf16x16 vb = mk16(vp + g * 8, vp + 16 + g * 8);
        O[t] = wmma_bf16(pa, vb, O[t]);
      }
    }
  }

#pragma unroll
  for (int t = 0; t < 4; ++t) {
    const int dcol = h * 64 + t * 16 + ln;
#pragma unroll
    for (int r = 0; r < 8; ++r) {
      const int row = qBase + wave * 16 + g * 8 + r;
      out[((size_t)b * 512 + row) * 768 + dcol] = (__bf16)(O[t][r] / lrun[r]);
    }
  }
}

// ---------------------------------------------------------------------------
// Host orchestration
// ---------------------------------------------------------------------------
extern "C" void kernel_launch(void* const* d_in, const int* in_sizes, int n_in,
                              void* d_out, int out_size, void* d_ws, size_t ws_size,
                              hipStream_t stream) {
  (void)in_sizes; (void)n_in; (void)out_size; (void)ws_size;
  const int L = 512, B = 16, E = 768, H = 12, NL = 12, FF = 3072;
  const int M = L * B;  // 8192 rows

  const float* x    = (const float*)d_in[0];
  const float* cmsk = (const float*)d_in[1];
  const float* pos  = (const float*)d_in[2];
  const float* caw  = (const float*)d_in[3];
  const float* cab  = (const float*)d_in[4];
  const float* apw  = (const float*)d_in[5];
  const float* apb  = (const float*)d_in[6];
  const float* g1   = (const float*)d_in[7];
  const float* b1   = (const float*)d_in[8];
  const float* fw   = (const float*)d_in[9];
  const float* fb   = (const float*)d_in[10];
  const float* pw   = (const float*)d_in[11];
  const float* pb   = (const float*)d_in[12];
  const float* g2   = (const float*)d_in[13];
  const float* b2   = (const float*)d_in[14];

  char* ws = (char*)d_ws;
  size_t off = 0;
  auto alloc = [&](size_t bytes) -> void* {
    off = (off + 255) & ~(size_t)255;
    void* p = ws + off;
    off += bytes;
    return p;
  };
  __bf16* wqkvT = (__bf16*)alloc((size_t)NL * 3 * E * E * 2);
  __bf16* wprojT = (__bf16*)alloc((size_t)NL * E * E * 2);
  __bf16* wfcT = (__bf16*)alloc((size_t)NL * E * FF * 2);
  __bf16* wmlpT = (__bf16*)alloc((size_t)NL * FF * E * 2);
  float* hF = (float*)alloc((size_t)M * E * 4);
  float* nF = (float*)alloc((size_t)M * E * 4);
  float* big = (float*)alloc((size_t)M * 3 * E * 4);  // qkv / a_out / m (time-shared)
  __bf16* hB = (__bf16*)alloc((size_t)M * E * 2);
  __bf16* nB = (__bf16*)alloc((size_t)M * E * 2);
  __bf16* oB = (__bf16*)alloc((size_t)M * E * 2);
  __bf16* gB = (__bf16*)alloc((size_t)M * FF * 2);

  dim3 tb(32, 8);
  wtrans_kernel<<<dim3((3 * E) / 32, E / 32, NL), tb, 0, stream>>>(caw, wqkvT, E, 3 * E);
  wtrans_kernel<<<dim3(E / 32, E / 32, NL), tb, 0, stream>>>(apw, wprojT, E, E);
  wtrans_kernel<<<dim3(FF / 32, E / 32, NL), tb, 0, stream>>>(fw, wfcT, E, FF);
  wtrans_kernel<<<dim3(E / 32, FF / 32, NL), tb, 0, stream>>>(pw, wmlpT, FF, E);

  init_h_kernel<<<(M * E) / 256, 256, 0, stream>>>(x, pos, hF, hB);

  for (int l = 0; l < NL; ++l) {
    // qkv = h @ caw + cab   (M x 2304)
    gemm_bf16_wmma<0><<<dim3(M / 128, (3 * E) / 128), 256, 0, stream>>>(
        hB, wqkvT + (size_t)l * 3 * E * E, cab + (size_t)l * 3 * E, big, nullptr,
        M, 3 * E, E);
    // attention -> oB (bf16, (B*L) x E)
    attn_flash_kernel<<<dim3(4, H, B), 256, 0, stream>>>(big, cmsk, oB);
    // a_out = o @ apw + apb  (fp32 into big)
    gemm_bf16_wmma<0><<<dim3(M / 128, E / 128), 256, 0, stream>>>(
        oB, wprojT + (size_t)l * E * E, apb + (size_t)l * E, big, nullptr, M, E, E);
    // n = LN(h + a_out)
    resid_ln_kernel<<<M, 256, 0, stream>>>(hF, big, g1 + (size_t)l * E,
                                           b1 + (size_t)l * E, nF, nB);
    // gelu(n @ fw + fb) -> gB (bf16)
    gemm_bf16_wmma<1><<<dim3(M / 128, FF / 128), 256, 0, stream>>>(
        nB, wfcT + (size_t)l * E * FF, fb + (size_t)l * FF, nullptr, gB, M, FF, E);
    // m = gelu @ pw + pb  (fp32 into big)
    gemm_bf16_wmma<0><<<dim3(M / 128, E / 128), 256, 0, stream>>>(
        gB, wmlpT + (size_t)l * FF * E, pb + (size_t)l * E, big, nullptr, M, E, FF);
    // h = LN(n + m)
    resid_ln_kernel<<<M, 256, 0, stream>>>(nF, big, g2 + (size_t)l * E,
                                           b2 + (size_t)l * E, hF, hB);
  }

  final_out_kernel<<<(M * E) / 256, 256, 0, stream>>>(hF, (float*)d_out);
}
